// _AxialMixtureAtt_50302656971163
// MI455X (gfx1250) — matrix operand order repaired and checked by reading, not compiled, not run on hardware
//
#include <hip/hip_runtime.h>
#include <hip/hip_bf16.h>

// ---------------------------------------------------------------------------
// AxialMixtureAtt for MI455X (gfx1250), wave32 + WMMA bf16 path.
// B=2, MAX_LEN=4096, D=1024, SEG=64, G=64, H=16, HD=64.
// All contractions run on v_wmma_f32_16x16x32_bf16; intermediates kept bf16
// (halves HBM traffic; 192MB L2 captures most of the ~330MB working set).
// Attention-side WMMA operands are staged through LDS (coalesced b128 global
// reads, ds_load fragment feeds); projection GEMMs use 16x64 strips per wave
// with all five fragments live so loads stay in flight across the four WMMAs.
// ---------------------------------------------------------------------------

typedef __bf16 bf16;
typedef __attribute__((ext_vector_type(16))) bf16  v16bf;
typedef __attribute__((ext_vector_type(8)))  float v8f;

#define NEGBIG (-1e10f)

// ---- WMMA fragment helpers (CDNA5 16x16x32 bf16 layouts, wave32) ----------

// A-matrix 16x32: lane L -> row L%16, hi=L/16; element e -> K=(e/8)*16+hi*8+e%8
__device__ __forceinline__ v16bf load_a(const bf16* base, int ld, int lane) {
  int r = lane & 15, hi = (lane >> 4) & 1;
  const bf16* p = base + r * ld + (hi << 3);
  v16bf out;
#pragma unroll
  for (int e = 0; e < 8; ++e) { out[e] = p[e]; out[e + 8] = p[16 + e]; }
  return out;
}
// A from f32 source (contiguous K), converted to bf16 on the fly
__device__ __forceinline__ v16bf load_a_f32(const float* base, int ld, int lane) {
  int r = lane & 15, hi = (lane >> 4) & 1;
  const float* p = base + r * ld + (hi << 3);
  v16bf out;
#pragma unroll
  for (int e = 0; e < 8; ++e) { out[e] = (bf16)p[e]; out[e + 8] = (bf16)p[16 + e]; }
  return out;
}
// B-matrix 32x16 where B[k][n] = src[n*ld + k]  (i.e. B = K^T, rows of K contiguous)
__device__ __forceinline__ v16bf load_bT(const bf16* base, int ld, int lane) {
  int n = lane & 15, hi = (lane >> 4) & 1;
  const bf16* p = base + n * ld + (hi << 3);
  v16bf out;
#pragma unroll
  for (int e = 0; e < 8; ++e) { out[e] = p[e]; out[e + 8] = p[16 + e]; }
  return out;
}

__device__ __forceinline__ v8f wmma_bf16(v16bf a, v16bf b, v8f c) {
  return __builtin_amdgcn_wmma_f32_16x16x32_bf16(false, a, false, b, (short)0, c,
                                                 false, false);
}

// C/D 16x16 f32: lane -> N=lane%16, hi=lane/16; VGPR r -> M = r + hi*8
__device__ __forceinline__ void store_c_f32(float* base, int ld, int lane, v8f c) {
  int n = lane & 15, hi = (lane >> 4) & 1;
#pragma unroll
  for (int r = 0; r < 8; ++r) base[(r + hi * 8) * ld + n] = c[r];
}
__device__ __forceinline__ void store_c_add_f32(float* base, int ld, int lane, v8f c) {
  int n = lane & 15, hi = (lane >> 4) & 1;
#pragma unroll
  for (int r = 0; r < 8; ++r) base[(r + hi * 8) * ld + n] += c[r];
}
__device__ __forceinline__ void store_c_bf16(bf16* base, int ld, int lane, v8f c,
                                             float scale) {
  int n = lane & 15, hi = (lane >> 4) & 1;
#pragma unroll
  for (int r = 0; r < 8; ++r) base[(r + hi * 8) * ld + n] = (bf16)(c[r] * scale);
}

// ---------------------------------------------------------------------------
// K0: f32 -> bf16 conversion of the input activations
__global__ void k_convert_x(const float* __restrict__ x, bf16* __restrict__ xb, int n) {
  int i = (blockIdx.x * blockDim.x + threadIdx.x) * 4;
  if (i + 3 < n) {
    float4 f = *(const float4*)(x + i);
    xb[i] = (bf16)f.x; xb[i + 1] = (bf16)f.y; xb[i + 2] = (bf16)f.z; xb[i + 3] = (bf16)f.w;
  }
}

// K0b: transpose+convert a 1024x1024 weight matrix: wt[n*1024+k] = w[k*1024+n]
__global__ void k_transpose_w(const float* __restrict__ w, bf16* __restrict__ wt) {
  int idx = blockIdx.x * blockDim.x + threadIdx.x;  // 0..1048575
  int k = idx >> 10, n = idx & 1023;
  wt[n * 1024 + k] = (bf16)w[idx];
}

// KA: QKV projection, M=8192 N=1024 K=1024 x3.
// One wave per 16x64 strip: one A fragment + four independent B fragments per
// k-step so VMEM stays ahead of the XDL pipe (partial loadcnt waits).
__global__ void k_qkv_proj(const bf16* __restrict__ xb, const bf16* __restrict__ wqt,
                           const bf16* __restrict__ wkt, const bf16* __restrict__ wvt,
                           bf16* __restrict__ q, bf16* __restrict__ kk,
                           bf16* __restrict__ v) {
  int lane = threadIdx.x, wave = threadIdx.y;
  int mt = blockIdx.x;                 // 0..511
  int ng = blockIdx.y * 4 + wave;      // 0..15 (n-strip of 64)
  int z = blockIdx.z;                  // 0=q,1=k,2=v
  const bf16* wt = (z == 0) ? wqt : (z == 1 ? wkt : wvt);
  bf16* dst = (z == 0) ? q : (z == 1 ? kk : v);
  const bf16* A  = xb + mt * 16 * 1024;
  const bf16* B0 = wt + ng * 64 * 1024;
  v8f a0 = {}, a1 = {}, a2 = {}, a3 = {};
#pragma unroll 2
  for (int k0 = 0; k0 < 1024; k0 += 32) {
    v16bf a  = load_a(A + k0, 1024, lane);
    v16bf b0 = load_bT(B0 + k0, 1024, lane);
    v16bf b1 = load_bT(B0 + 16 * 1024 + k0, 1024, lane);
    v16bf b2 = load_bT(B0 + 32 * 1024 + k0, 1024, lane);
    v16bf b3 = load_bT(B0 + 48 * 1024 + k0, 1024, lane);
    a0 = wmma_bf16(a, b0, a0);
    a1 = wmma_bf16(a, b1, a1);
    a2 = wmma_bf16(a, b2, a2);
    a3 = wmma_bf16(a, b3, a3);
  }
  float sc = (z == 0) ? 0.125f : 1.0f;  // q pre-scaled by 1/sqrt(HD)=1/8
  bf16* d = dst + mt * 16 * 1024 + ng * 64;
  store_c_bf16(d, 1024, lane, a0, sc);
  store_c_bf16(d + 16, 1024, lane, a1, sc);
  store_c_bf16(d + 32, 1024, lane, a2, sc);
  store_c_bf16(d + 48, 1024, lane, a3, sc);
}

// KB: cummax scans. cfg(blockIdx.y): 0=fwd over segments, 1=fwd over pos, 2=rev over pos.
// q-variant is unshifted cummax; k-variant is cummax shifted by one (pad=0).
__global__ void k_cummax(const bf16* __restrict__ qin, const bf16* __restrict__ kin,
                         bf16* __restrict__ cuq, bf16* __restrict__ cuk,
                         bf16* __restrict__ rlq, bf16* __restrict__ rlk,
                         bf16* __restrict__ rrq, bf16* __restrict__ rrk) {
  int cfg = blockIdx.y;
  int tid = blockIdx.x * blockDim.x + threadIdx.x;  // 131072 chains
  int b = tid >> 16;
  int base, stride;
  if (cfg == 0) { base = b * 4194304 + (tid & 65535); stride = 65536; }
  else { int g = (tid >> 10) & 63; int c = tid & 1023;
         base = b * 4194304 + g * 65536 + c; stride = 1024; }
  bf16* qo = (cfg == 0) ? cuq : (cfg == 1 ? rlq : rrq);
  bf16* ko = (cfg == 0) ? cuk : (cfg == 1 ? rlk : rrk);
  float rq = -3.0e38f, rk = -3.0e38f;
  if (cfg < 2) {
    for (int i = 0; i < 64; ++i) {
      int o = base + i * stride;
      ko[o] = (i == 0) ? (bf16)0.0f : (bf16)rk;
      rk = fmaxf(rk, (float)kin[o]);
      rq = fmaxf(rq, (float)qin[o]);
      qo[o] = (bf16)rq;
    }
  } else {
    for (int i = 63; i >= 0; --i) {
      int o = base + i * stride;
      ko[o] = (i == 63) ? (bf16)0.0f : (bf16)rk;
      rk = fmaxf(rk, (float)kin[o]);
      rq = fmaxf(rq, (float)qin[o]);
      qo[o] = (bf16)rq;
    }
  }
}

// KC: the three 64x64 softmax-attention summaries (2048 problems per mode).
// mode(blockIdx.y): 0=col_u2d (batch over pos), 1=row_l2r, 2=row_r2l (batch over seg)
// Q/K staged natural, V staged transposed in LDS; all WMMA operands come from LDS.
__global__ void k_summary(const bf16* __restrict__ cuq, const bf16* __restrict__ rlq,
                          const bf16* __restrict__ rrq, const bf16* __restrict__ karr,
                          const bf16* __restrict__ varr, bf16* __restrict__ colsum,
                          bf16* __restrict__ rlsum, bf16* __restrict__ rrsum) {
  __shared__ float sS[64][68];
  __shared__ bf16  sP[64][64];
  __shared__ bf16  sQ[64][72];
  __shared__ bf16  sK[64][72];
  __shared__ bf16  sVt[64][72];  // transposed: sVt[hd][t]
  int mode = blockIdx.y;
  int lane = threadIdx.x & 31, wave = threadIdx.x >> 5;
  int x = blockIdx.x;
  int b = x >> 10, idx = (x >> 4) & 63, h = x & 15;
  int boff, ld;
  if (mode == 0) { boff = b * 4194304 + idx * 1024 + h * 64;  ld = 65536; }
  else           { boff = b * 4194304 + idx * 65536 + h * 64; ld = 1024;  }
  const bf16* Qb = ((mode == 0) ? cuq : (mode == 1) ? rlq : rrq) + boff;
  const bf16* Kb = karr + boff;
  const bf16* Vb = varr + boff;
  bf16* Db = ((mode == 0) ? colsum : (mode == 1) ? rlsum : rrsum) + boff;

  // cooperative stage (coalesced 128B row reads)
  for (int e = threadIdx.x; e < 64 * 64; e += 256) {
    int r = e >> 6, c = e & 63;
    sQ[r][c]  = Qb[r * ld + c];
    sK[r][c]  = Kb[r * ld + c];
    sVt[c][r] = Vb[r * ld + c];
  }
  __syncthreads();

  // S = Q K^T  (16 tiles / 8 waves)
  for (int i = 0; i < 2; ++i) {
    int t = wave * 2 + i, ti = t >> 2, tj = t & 3;
    v8f acc = {};
#pragma unroll
    for (int k0 = 0; k0 < 64; k0 += 32)
      acc = wmma_bf16(load_a(&sQ[ti * 16][k0], 72, lane),
                      load_bT(&sK[tj * 16][k0], 72, lane), acc);
    store_c_f32(&sS[ti * 16][tj * 16], 68, lane, acc);
  }
  __syncthreads();

  // masked softmax per row (mode 0/1: t<=s allowed; mode 2: t>=s allowed)
  if (threadIdx.x < 64) {
    int s = threadIdx.x;
    float mx = -3.0e38f;
    for (int tt = 0; tt < 64; ++tt) {
      bool ok = (mode == 2) ? (tt >= s) : (tt <= s);
      float l = ok ? sS[s][tt] : NEGBIG;
      sS[s][tt] = l;
      mx = fmaxf(mx, l);
    }
    float sum = 0.f;
    for (int tt = 0; tt < 64; ++tt) { float e = __expf(sS[s][tt] - mx); sum += e; sS[s][tt] = e; }
    float inv = 1.0f / sum;
    for (int tt = 0; tt < 64; ++tt) sP[s][tt] = (bf16)(sS[s][tt] * inv);
  }
  __syncthreads();

  // O = P V (both operands from LDS)
  for (int i = 0; i < 2; ++i) {
    int t = wave * 2 + i, ti = t >> 2, tj = t & 3;
    v8f acc = {};
#pragma unroll
    for (int k0 = 0; k0 < 64; k0 += 32)
      acc = wmma_bf16(load_a(&sP[ti * 16][k0], 64, lane),
                      load_bT(&sVt[tj * 16][k0], 72, lane), acc);
    store_c_f32(&sS[ti * 16][tj * 16], 68, lane, acc);
  }
  __syncthreads();

  // shifted store: modes 0/1 -> row s stored at s+1, row 0 zeroed;
  //                mode 2    -> row s stored at s-1, row 63 zeroed.
  for (int e = threadIdx.x; e < 64 * 64; e += 256) {
    int s = e >> 6, d = e & 63;
    int sd = (mode == 2) ? s - 1 : s + 1;
    if (sd >= 0 && sd < 64) Db[sd * ld + d] = (bf16)sS[s][d];
  }
  if (threadIdx.x < 64) {
    int zrow = (mode == 2) ? 63 : 0;
    Db[zrow * ld + threadIdx.x] = (bf16)0.0f;
  }
}

// KD1: col / top-left / top-right logit blocks, batched per (b, pos u, head).
// Writes bf16 logits (+bias) at comp offsets 0 / 192 / 256. Q+3K staged in LDS.
__global__ void k_logits_sut(const bf16* __restrict__ q, const bf16* __restrict__ k,
                             const bf16* __restrict__ rlk, const bf16* __restrict__ rrk,
                             bf16* __restrict__ lg) {
  __shared__ bf16 sQ[64][72];
  __shared__ bf16 sK[3][64][72];
  int lane = threadIdx.x & 31, wave = threadIdx.x >> 5;
  int x = blockIdx.x;
  int b = x >> 10, u = (x >> 4) & 63, h = x & 15;
  int boff = b * 4194304 + u * 1024 + h * 64;
  const bf16* Qb = q + boff;
  const bf16* K0 = k + boff;
  const bf16* K1 = rlk + boff;
  const bf16* K2 = rrk + boff;
  bf16* Lb = lg + ((b * 16 + h) * 4096 + u) * 320;  // row stride over s(seg) = 20480
  const int  cofs[3] = { 0, 192, 256 };
  const bool en[3]   = { true, u != 0, u != 63 };

  for (int e = threadIdx.x; e < 64 * 64; e += 256) {
    int r = e >> 6, c = e & 63;
    int o = r * 65536 + c;
    sQ[r][c] = Qb[o]; sK[0][r][c] = K0[o]; sK[1][r][c] = K1[o]; sK[2][r][c] = K2[o];
  }
  __syncthreads();

  for (int i = 0; i < 6; ++i) {
    int t = wave * 6 + i;
    int mat = t >> 4, w16 = t & 15, ti = w16 >> 2, tj = w16 & 3;
    v8f acc = {};
#pragma unroll
    for (int k0 = 0; k0 < 64; k0 += 32)
      acc = wmma_bf16(load_a(&sQ[ti * 16][k0], 72, lane),
                      load_bT(&sK[mat][tj * 16][k0], 72, lane), acc);
    int n = lane & 15, hi = (lane >> 4) & 1;
#pragma unroll
    for (int r = 0; r < 8; ++r) {
      int s = ti * 16 + hi * 8 + r, tt = tj * 16 + n;
      bool ok = en[mat] && (tt < s);  // strict causal over segments
      Lb[s * 20480 + cofs[mat] + tt] = (bf16)(ok ? acc[r] : NEGBIG);
    }
  }
}

// KD2: row / previous-row logit blocks, batched per (b, seg g, head).
// comp offsets 64 (row: t<=s) / 128 (prev: g!=0 && t!=s). Q+2K staged in LDS.
__global__ void k_logits_row(const bf16* __restrict__ q, const bf16* __restrict__ k,
                             const bf16* __restrict__ cuk, bf16* __restrict__ lg) {
  __shared__ bf16 sQ[64][72];
  __shared__ bf16 sK[2][64][72];
  int lane = threadIdx.x & 31, wave = threadIdx.x >> 5;
  int x = blockIdx.x;
  int b = x >> 10, g = (x >> 4) & 63, h = x & 15;
  int boff = b * 4194304 + g * 65536 + h * 64;
  const bf16* Qb = q + boff;
  const bf16* K0 = k + boff;
  const bf16* K1 = cuk + boff;
  bf16* Lb = lg + ((b * 16 + h) * 64 + g) * 20480;  // row stride over s(pos) = 320
  const int cofs[2] = { 64, 128 };

  for (int e = threadIdx.x; e < 64 * 64; e += 256) {
    int r = e >> 6, c = e & 63;
    int o = r * 1024 + c;
    sQ[r][c] = Qb[o]; sK[0][r][c] = K0[o]; sK[1][r][c] = K1[o];
  }
  __syncthreads();

  for (int i = 0; i < 4; ++i) {
    int t = wave * 4 + i;
    int mat = t >> 4, w16 = t & 15, ti = w16 >> 2, tj = w16 & 3;
    v8f acc = {};
#pragma unroll
    for (int k0 = 0; k0 < 64; k0 += 32)
      acc = wmma_bf16(load_a(&sQ[ti * 16][k0], 72, lane),
                      load_bT(&sK[mat][tj * 16][k0], 72, lane), acc);
    int n = lane & 15, hi = (lane >> 4) & 1;
#pragma unroll
    for (int r = 0; r < 8; ++r) {
      int s = ti * 16 + hi * 8 + r, tt = tj * 16 + n;
      bool ok = (mat == 0) ? (tt <= s) : (g != 0 && tt != s);
      Lb[s * 320 + cofs[mat] + tt] = (bf16)(ok ? acc[r] : NEGBIG);
    }
  }
}

// KE0: joint softmax over 320 mixture logits, one wave per query row.
__global__ void k_softmax(bf16* __restrict__ lg) {
  int lane = threadIdx.x & 31, wave = threadIdx.x >> 5;
  int row = blockIdx.x * 8 + wave;  // < 131072
  bf16* p = lg + (long)row * 320;
  float vals[10];
  float mx = -3.0e38f;
#pragma unroll
  for (int i = 0; i < 10; ++i) { vals[i] = (float)p[lane + i * 32]; mx = fmaxf(mx, vals[i]); }
#pragma unroll
  for (int d = 16; d; d >>= 1) mx = fmaxf(mx, __shfl_xor(mx, d, 32));
  float sum = 0.f;
#pragma unroll
  for (int i = 0; i < 10; ++i) { vals[i] = __expf(vals[i] - mx); sum += vals[i]; }
#pragma unroll
  for (int d = 16; d; d >>= 1) sum += __shfl_xor(sum, d, 32);
  float inv = 1.0f / sum;
#pragma unroll
  for (int i = 0; i < 10; ++i) p[lane + i * 32] = (bf16)(vals[i] * inv);
}

// KE1: col_m + tl_m + tr_m mixtures (batched per (b, pos u, head)); writes merged.
// The three 64x64 B-matrices are staged transposed in LDS (coalesced reads, 4x reuse).
__global__ void k_mix_sut(const bf16* __restrict__ lg, const bf16* __restrict__ v,
                          const bf16* __restrict__ rls, const bf16* __restrict__ rrs,
                          float* __restrict__ merged) {
  __shared__ bf16 sB[3][64][72];  // sB[m][hd][t]
  int lane = threadIdx.x & 31, wave = threadIdx.x >> 5;
  int x = blockIdx.x;
  int b = x >> 10, u = (x >> 4) & 63, h = x & 15;
  int boff = b * 4194304 + u * 1024 + h * 64;
  const bf16* Ab  = lg + ((b * 16 + h) * 4096 + u) * 320;  // rows = seg s, ld 20480
  const bf16* Vb  = v + boff;
  const bf16* TLb = rls + boff;
  const bf16* TRb = rrs + boff;
  float* Mb = merged + boff;      // rows = seg s, ld 65536

  for (int e = threadIdx.x; e < 64 * 64; e += 256) {
    int t = e >> 6, n = e & 63;
    int o = t * 65536 + n;
    sB[0][n][t] = Vb[o]; sB[1][n][t] = TLb[o]; sB[2][n][t] = TRb[o];
  }
  __syncthreads();

  for (int i = 0; i < 2; ++i) {
    int t = wave * 2 + i, ti = t >> 2, tj = t & 3;
    const bf16* Ar = Ab + ti * 16 * 20480;
    v8f acc = {};
#pragma unroll
    for (int k0 = 0; k0 < 64; k0 += 32) {
      acc = wmma_bf16(load_a(Ar + k0, 20480, lane),
                      load_bT(&sB[0][tj * 16][k0], 72, lane), acc);
      acc = wmma_bf16(load_a(Ar + 192 + k0, 20480, lane),
                      load_bT(&sB[1][tj * 16][k0], 72, lane), acc);
      acc = wmma_bf16(load_a(Ar + 256 + k0, 20480, lane),
                      load_bT(&sB[2][tj * 16][k0], 72, lane), acc);
    }
    store_c_f32(Mb + ti * 16 * 65536 + tj * 16, 65536, lane, acc);
  }
}

// KE2: row_m + prev_m mixtures (batched per (b, seg g, head)); accumulates into merged.
__global__ void k_mix_row(const bf16* __restrict__ lg, const bf16* __restrict__ v,
                          const bf16* __restrict__ cs, float* __restrict__ merged) {
  __shared__ bf16 sB[2][64][72];  // sB[m][hd][t]
  int lane = threadIdx.x & 31, wave = threadIdx.x >> 5;
  int x = blockIdx.x;
  int b = x >> 10, g = (x >> 4) & 63, h = x & 15;
  int boff = b * 4194304 + g * 65536 + h * 64;
  const bf16* Ab = lg + ((b * 16 + h) * 64 + g) * 20480;  // rows = pos s, ld 320
  const bf16* Vb = v + boff;
  const bf16* Cb = cs + boff;
  float* Mb = merged + boff;   // ld 1024

  for (int e = threadIdx.x; e < 64 * 64; e += 256) {
    int t = e >> 6, n = e & 63;
    int o = t * 1024 + n;
    sB[0][n][t] = Vb[o]; sB[1][n][t] = Cb[o];
  }
  __syncthreads();

  for (int i = 0; i < 2; ++i) {
    int t = wave * 2 + i, ti = t >> 2, tj = t & 3;
    const bf16* Ar = Ab + ti * 16 * 320;
    v8f acc = {};
#pragma unroll
    for (int k0 = 0; k0 < 64; k0 += 32) {
      acc = wmma_bf16(load_a(Ar + 64 + k0, 320, lane),
                      load_bT(&sB[0][tj * 16][k0], 72, lane), acc);
      acc = wmma_bf16(load_a(Ar + 128 + k0, 320, lane),
                      load_bT(&sB[1][tj * 16][k0], 72, lane), acc);
    }
    store_c_add_f32(Mb + ti * 16 * 1024 + tj * 16, 1024, lane, acc);
  }
}

// KF: output projection merged[8192x1024] x WoT -> out f32 (16x64 strip per wave)
__global__ void k_out_proj(const float* __restrict__ merged, const bf16* __restrict__ wot,
                           float* __restrict__ out) {
  int lane = threadIdx.x, wave = threadIdx.y;
  int mt = blockIdx.x;
  int ng = blockIdx.y * 4 + wave;  // 0..15
  const float* A = merged + mt * 16 * 1024;
  const bf16* B0 = wot + ng * 64 * 1024;
  v8f a0 = {}, a1 = {}, a2 = {}, a3 = {};
#pragma unroll 2
  for (int k0 = 0; k0 < 1024; k0 += 32) {
    v16bf a  = load_a_f32(A + k0, 1024, lane);
    v16bf b0 = load_bT(B0 + k0, 1024, lane);
    v16bf b1 = load_bT(B0 + 16 * 1024 + k0, 1024, lane);
    v16bf b2 = load_bT(B0 + 32 * 1024 + k0, 1024, lane);
    v16bf b3 = load_bT(B0 + 48 * 1024 + k0, 1024, lane);
    a0 = wmma_bf16(a, b0, a0);
    a1 = wmma_bf16(a, b1, a1);
    a2 = wmma_bf16(a, b2, a2);
    a3 = wmma_bf16(a, b3, a3);
  }
  float* d = out + mt * 16 * 1024 + ng * 64;
  store_c_f32(d, 1024, lane, a0);
  store_c_f32(d + 16, 1024, lane, a1);
  store_c_f32(d + 32, 1024, lane, a2);
  store_c_f32(d + 48, 1024, lane, a3);
}

// ---------------------------------------------------------------------------
extern "C" void kernel_launch(void* const* d_in, const int* in_sizes, int n_in,
                              void* d_out, int out_size, void* d_ws, size_t ws_size,
                              hipStream_t stream) {
  (void)in_sizes; (void)n_in; (void)out_size; (void)ws_size;
  const float* x  = (const float*)d_in[0];
  const float* Wq = (const float*)d_in[1];
  const float* Wk = (const float*)d_in[2];
  const float* Wv = (const float*)d_in[3];
  const float* Wo = (const float*)d_in[4];
  float* out = (float*)d_out;

  char* w = (char*)d_ws;
  auto alloc = [&](size_t bytes) -> void* {
    void* p = (void*)w; w += (bytes + 255) & ~(size_t)255; return p;
  };
  const size_t NE = 2ull * 4096 * 1024;  // 8,388,608 activation elements
  bf16* xb   = (bf16*)alloc(NE * 2);
  bf16* wqt  = (bf16*)alloc(1024ull * 1024 * 2);
  bf16* wkt  = (bf16*)alloc(1024ull * 1024 * 2);
  bf16* wvt  = (bf16*)alloc(1024ull * 1024 * 2);
  bf16* wot  = (bf16*)alloc(1024ull * 1024 * 2);
  bf16* qb   = (bf16*)alloc(NE * 2);
  bf16* kb   = (bf16*)alloc(NE * 2);
  bf16* vb   = (bf16*)alloc(NE * 2);
  bf16* cuq  = (bf16*)alloc(NE * 2);   // cummax(q) over segments
  bf16* cuk  = (bf16*)alloc(NE * 2);   // shifted cummax(k) over segments
  bf16* rlq  = (bf16*)alloc(NE * 2);   // l2r cummax(q) over positions
  bf16* rlk  = (bf16*)alloc(NE * 2);
  bf16* rrq  = (bf16*)alloc(NE * 2);   // r2l cummax(q) over positions
  bf16* rrk  = (bf16*)alloc(NE * 2);
  bf16* csum = (bf16*)alloc(NE * 2);   // col_u2d_summary
  bf16* rlsm = (bf16*)alloc(NE * 2);   // row_l2r_summary
  bf16* rrsm = (bf16*)alloc(NE * 2);   // row_r2l_summary
  bf16* lg   = (bf16*)alloc(131072ull * 320 * 2);  // joint logits/probs
  float* mg  = (float*)alloc(NE * 4);  // merged mixture output

  k_convert_x<<<8192, 256, 0, stream>>>(x, xb, (int)NE);
  k_transpose_w<<<4096, 256, 0, stream>>>(Wq, wqt);
  k_transpose_w<<<4096, 256, 0, stream>>>(Wk, wkt);
  k_transpose_w<<<4096, 256, 0, stream>>>(Wv, wvt);
  k_transpose_w<<<4096, 256, 0, stream>>>(Wo, wot);

  k_qkv_proj<<<dim3(512, 4, 3), dim3(32, 4), 0, stream>>>(xb, wqt, wkt, wvt, qb, kb, vb);
  k_cummax<<<dim3(512, 3), 256, 0, stream>>>(qb, kb, cuq, cuk, rlq, rlk, rrq, rrk);
  k_summary<<<dim3(2048, 3), 256, 0, stream>>>(cuq, rlq, rrq, kb, vb, csum, rlsm, rrsm);
  k_logits_sut<<<2048, 256, 0, stream>>>(qb, kb, rlk, rrk, lg);
  k_logits_row<<<2048, 256, 0, stream>>>(qb, kb, cuk, lg);
  k_softmax<<<16384, 256, 0, stream>>>(lg);
  k_mix_sut<<<2048, 256, 0, stream>>>(lg, vb, rlsm, rrsm, mg);
  k_mix_row<<<2048, 256, 0, stream>>>(lg, vb, csum, mg);
  k_out_proj<<<dim3(512, 4), dim3(32, 4), 0, stream>>>(mg, wot, out);
}
